// GCN_35218731827634
// MI455X (gfx1250) — matrix operand order repaired and checked by reading, not compiled, not run on hardware
//
#include <hip/hip_runtime.h>
#include <hip/hip_bf16.h>

#define N_NODES 100000
#define N_EDGES 640000
#define D 128
#define NEG_SLOPE 0.01f

typedef __attribute__((ext_vector_type(16))) __bf16 v16bf;
typedef __attribute__((ext_vector_type(8)))  float  v8f;

// ---------------- degree / normalization ----------------

__global__ void gcn_deg_init(float* deg) {
    int i = blockIdx.x * 256 + threadIdx.x;
    if (i < N_NODES) deg[i] = 1.0f;   // self loop
}

__global__ void gcn_deg_accum(const int* __restrict__ dst, float* deg) {
    int e = blockIdx.x * 256 + threadIdx.x;
    if (e < N_EDGES) atomicAdd(&deg[dst[e]], 1.0f);
}

__global__ void gcn_dinv(float* deg) {
    int i = blockIdx.x * 256 + threadIdx.x;
    if (i < N_NODES) deg[i] = rsqrtf(deg[i]);   // deg >= 1 always
}

// ---------------- W (f32 row-major [k][n]) -> Wt (bf16 [n][k]) ----------------

__global__ void gcn_wconv(const float* __restrict__ W, __bf16* __restrict__ Wt) {
    int t = blockIdx.x * 256 + threadIdx.x;   // 16384
    if (t >= D * D) return;
    int k = t >> 7, n = t & (D - 1);
    Wt[n * D + k] = (__bf16)W[k * D + n];
}

// ---------------- H = leaky_relu(X) @ W  via v_wmma_f32_16x16x32_bf16 ----------------
// block = 256 threads (8 waves); each wave computes a 16x128 strip of H.

__global__ __launch_bounds__(256) void gcn_gemm_act(const float* __restrict__ X,
                                                    const __bf16* __restrict__ Wt,
                                                    float* __restrict__ H) {
    const int lane = threadIdx.x & 31;
    const int wave = threadIdx.x >> 5;
    const int half = lane >> 4;      // 0: lanes 0-15, 1: lanes 16-31
    const int l16  = lane & 15;

    const int  mBase = blockIdx.x * 128 + wave * 16;
    int        row   = mBase + l16;
    const int  rclmp = (row < N_NODES) ? row : (N_NODES - 1);   // keep EXEC all-1s for WMMA
    const float* xrow = X + (long)rclmp * D;

    v8f acc[8] = {};   // 8 column tiles of 16x16 f32

    #pragma unroll
    for (int kt = 0; kt < 4; ++kt) {
        const int kb = kt * 32;
        const int ka = kb + half * 8;     // A layout: lanes0-15 K {0..7,16..23}, lanes16-31 K {8..15,24..31}
        v16bf a;
        #pragma unroll
        for (int i = 0; i < 8; ++i) {
            float v0 = xrow[ka + i];
            float v1 = xrow[ka + 16 + i];
            v0 = (v0 > 0.0f) ? v0 : NEG_SLOPE * v0;   // fused LeakyReLU on A
            v1 = (v1 > 0.0f) ? v1 : NEG_SLOPE * v1;
            a[i]     = (__bf16)v0;
            a[8 + i] = (__bf16)v1;
        }
        #pragma unroll
        for (int ct = 0; ct < 8; ++ct) {
            const int col = ct * 16 + l16;
            // B layout: lanes0-15 K {0..7,16..23}, lanes16-31 K {8..15,24..31} (matches A)
            const __bf16* wt = Wt + col * D + kb + half * 8;
            v16bf b;
            #pragma unroll
            for (int i = 0; i < 16; ++i) b[i] = wt[i < 8 ? i : i + 8];
            acc[ct] = __builtin_amdgcn_wmma_f32_16x16x32_bf16(
                false, a, false, b, (short)0, acc[ct], false, false);
        }
    }

    // C/D layout: component v -> M = v + 8*half, N = l16
    #pragma unroll
    for (int ct = 0; ct < 8; ++ct) {
        const int col = ct * 16 + l16;
        #pragma unroll
        for (int v = 0; v < 8; ++v) {
            int r = mBase + v + half * 8;
            if (r < N_NODES) H[(long)r * D + col] = acc[ct][v];
        }
    }
}

// ---------------- out = H * dinv^2 + bias (self loop term + bias, also zero-init) ----------------

__global__ void gcn_self_bias(const float* __restrict__ H, const float* __restrict__ dinv,
                              const float* __restrict__ bias, float* __restrict__ out) {
    long t = (long)blockIdx.x * 256 + threadIdx.x;
    if (t >= (long)N_NODES * D) return;
    int i = (int)(t >> 7), j = (int)(t & (D - 1));
    float di = dinv[i];
    out[t] = H[t] * di * di + bias[j];
}

// ---------------- edge scatter: out[d] += H[s] * dinv[s]*dinv[d]; one wave per edge ----------------

__global__ void gcn_edge_agg(const int* __restrict__ src, const int* __restrict__ dst,
                             const float* __restrict__ H, const float* __restrict__ dinv,
                             float* __restrict__ out) {
    long t = (long)blockIdx.x * 256 + threadIdx.x;
    int  e = (int)(t >> 5);
    int  lane = (int)(t & 31);
    if (e >= N_EDGES) return;
    int s = src[e], d = dst[e];
    float w = dinv[s] * dinv[d];
    const float4 v = ((const float4*)(H + (long)s * D))[lane];
    float* od = out + (long)d * D + lane * 4;
    atomicAdd(od + 0, v.x * w);
    atomicAdd(od + 1, v.y * w);
    atomicAdd(od + 2, v.z * w);
    atomicAdd(od + 3, v.w * w);
}

// ---------------- driver ----------------

extern "C" void kernel_launch(void* const* d_in, const int* in_sizes, int n_in,
                              void* d_out, int out_size, void* d_ws, size_t ws_size,
                              hipStream_t stream) {
    (void)in_sizes; (void)n_in; (void)out_size; (void)ws_size;

    const float* x   = (const float*)d_in[0];
    const int*   ei  = (const int*)d_in[1];      // [2, E] flat: src then dst
    const int*   src = ei;
    const int*   dst = ei + N_EDGES;
    const float* Ws[3] = { (const float*)d_in[2], (const float*)d_in[4], (const float*)d_in[6] };
    const float* bs[3] = { (const float*)d_in[3], (const float*)d_in[5], (const float*)d_in[7] };

    // workspace layout
    char* ws = (char*)d_ws;
    float*  hbuf = (float*)ws;                          ws += (size_t)N_NODES * D * sizeof(float);
    float*  agg  = (float*)ws;                          ws += (size_t)N_NODES * D * sizeof(float);
    float*  dinv = (float*)ws;                          ws += (size_t)N_NODES * sizeof(float);
    __bf16* Wt   = (__bf16*)ws;

    const int nodeBlocks = (N_NODES + 255) / 256;
    const int edgeBlocks = (N_EDGES + 255) / 256;
    const int gemmBlocks = (N_NODES + 127) / 128;             // 782
    const int elemBlocks = (int)(((long)N_NODES * D + 255) / 256);   // 50000
    const int scatBlocks = (int)(((long)N_EDGES * 32 + 255) / 256);  // 80000

    // degrees -> dinv (shared by all 3 layers)
    gcn_deg_init <<<nodeBlocks, 256, 0, stream>>>(dinv);
    gcn_deg_accum<<<edgeBlocks, 256, 0, stream>>>(dst, dinv);
    gcn_dinv     <<<nodeBlocks, 256, 0, stream>>>(dinv);

    const float* xin = x;
    for (int l = 0; l < 3; ++l) {
        float* outb = (l == 2) ? (float*)d_out : agg;
        gcn_wconv    <<<(D * D + 255) / 256, 256, 0, stream>>>(Ws[l], Wt);
        gcn_gemm_act <<<gemmBlocks, 256, 0, stream>>>(xin, Wt, hbuf);
        gcn_self_bias<<<elemBlocks, 256, 0, stream>>>(hbuf, dinv, bs[l], outb);
        gcn_edge_agg <<<scatBlocks, 256, 0, stream>>>(src, dst, hbuf, dinv, outb);
        xin = outb;   // gemm fully consumes xin before outb is rewritten next layer
    }
}
